// SelfAttention_25692494364785
// MI455X (gfx1250) — matrix operand order repaired and checked by reading, not compiled, not run on hardware
//
#include <hip/hip_runtime.h>

// ---------------------------------------------------------------------------
// Causal multi-head self-attention for MI455X (gfx1250), wave32 + WMMA bf16.
// B=4, W=2048, C=768, H=3, head_dim=256.
// Pipeline: f32->bf16 convert, QKV WMMA GEMM (async-LDS double buffered),
// flash-attention (WMMA, L2-fed), proj GEMM (async-LDS double buffered).
// ---------------------------------------------------------------------------

typedef __bf16 bf16;
typedef __attribute__((ext_vector_type(8)))  __bf16 v8bf;
typedef __attribute__((ext_vector_type(16))) __bf16 v16bf;
typedef __attribute__((ext_vector_type(8)))  float  v8f;

#define B_DIM 4
#define W_LEN 2048
#define C_DIM 768
#define H_DIM 3
#define K_DIM 256               // head dim
#define M_ROWS (B_DIM * W_LEN)  // 8192

// ---------------- WMMA helpers (layouts per CDNA5 ISA 7.12.2) ----

__device__ __forceinline__ v8f wmma_bf16(v16bf a, v16bf b, v8f c) {
  return __builtin_amdgcn_wmma_f32_16x16x32_bf16(
      false, a, false, b, (short)0, c, false, false);
}

// A-fragment 16x32 (M x K): lane holds row M=lane&15.
// elements 0..7  = K  8*half..   (at p);  8..15 = K 16+8*half.. (at p+16)
__device__ __forceinline__ v16bf load_frag_a(const bf16* p) {
  v8bf lo = *(const v8bf*)(p);
  v8bf hi = *(const v8bf*)(p + 16);
  v16bf r;
#pragma unroll
  for (int i = 0; i < 8; ++i) { r[i] = lo[i]; r[i + 8] = hi[i]; }
  return r;
}

// B-fragment 32x16 (K x N): lane holds column N=lane&15.
// elements 0..7  = K  8*half..   (at p);  8..15 = K 16+8*half.. (at p+16)
__device__ __forceinline__ v16bf load_frag_b(const bf16* p) {
  v8bf lo = *(const v8bf*)(p);
  v8bf hi = *(const v8bf*)(p + 16);
  v16bf r;
#pragma unroll
  for (int i = 0; i < 8; ++i) { r[i] = lo[i]; r[i + 8] = hi[i]; }
  return r;
}

// ---------------- CDNA5 async global->LDS copy (ASYNCcnt path) -------------
// global_load_async_to_lds_b128 vdst(lds addr), vaddr(64b global), off
// Generic pointers to __shared__ carry the LDS byte offset in addr[31:0]
// (flat->LDS mapping truncates), so casting gives the DS address.

__device__ __forceinline__ void async_copy_b128(void* lds_ptr,
                                                const void* gptr) {
  unsigned lds_addr = (unsigned)(unsigned long long)lds_ptr;
  asm volatile("global_load_async_to_lds_b128 %0, %1, off"
               :
               : "v"(lds_addr), "v"(gptr)
               : "memory");
}

__device__ __forceinline__ void async_wait0() {
  asm volatile("s_wait_asynccnt 0" ::: "memory");
}

// ---------------- f32 -> bf16 conversion ------------------------------------

__global__ void cvt_f32_bf16_kernel(const float4* __restrict__ src,
                                    bf16* __restrict__ dst, int n4) {
  int i = blockIdx.x * blockDim.x + threadIdx.x;
  int stride = gridDim.x * blockDim.x;
  for (; i < n4; i += stride) {
    float4 v = src[i];
    bf16* d = dst + i * 4;
    d[0] = (bf16)v.x; d[1] = (bf16)v.y; d[2] = (bf16)v.z; d[3] = (bf16)v.w;
  }
}

// ---------------- QKV projection GEMM ---------------------------------------
// D[m][d] = sum_k X[m][k] * Wqkv[d][k]   (M=8192, N=2304, K=768)
// Double-buffered LDS tiles filled by async copies; epilogue scatters into
// Q (scaled by 1/16), K, and V-transposed buffers.

#define GP 40  // LDS tile pitch in bf16 (32 + 8 pad), rows stay 16B-aligned

__global__ __launch_bounds__(256) void qkv_gemm_kernel(
    const bf16* __restrict__ X,    // [8192][768]
    const bf16* __restrict__ Wq,   // [2304][768]
    bf16* __restrict__ qb,         // [B*H][2048][256]  (pre-scaled)
    bf16* __restrict__ kb,         // [B*H][2048][256]
    bf16* __restrict__ vt) {       // [B*H][256][2048]  (transposed V)
  __shared__ bf16 lA[2][128 * GP];
  __shared__ bf16 lB[2][128 * GP];

  const int tid  = threadIdx.x;
  const int lane = tid & 31, wv = tid >> 5;
  const int half = lane >> 4, l16 = lane & 15;
  const int m0 = blockIdx.x * 128;
  const int n0 = blockIdx.y * 128;
  const int wm = (wv & 3) * 32;   // wave row offset (2 tiles of 16)
  const int wn = (wv >> 2) * 64;  // wave col offset (4 tiles of 16)

  v8f acc[2][4];
#pragma unroll
  for (int i = 0; i < 2; ++i)
#pragma unroll
    for (int j = 0; j < 4; ++j) acc[i][j] = (v8f)(0.0f);

  const int lrow = tid >> 1;        // 0..127
  const int lcol = (tid & 1) * 16;  // 0 or 16
  const bf16* gA = X  + (long)(m0 + lrow) * C_DIM + lcol;
  const bf16* gB = Wq + (long)(n0 + lrow) * C_DIM + lcol;
  const int lofs = lrow * GP + lcol;

  const int T = C_DIM / 32;  // 24 K-steps

  // prologue: tile 0 into buffer 0
  async_copy_b128(&lA[0][lofs],     gA);
  async_copy_b128(&lA[0][lofs + 8], gA + 8);
  async_copy_b128(&lB[0][lofs],     gB);
  async_copy_b128(&lB[0][lofs + 8], gB + 8);
  async_wait0();
  __syncthreads();

  for (int t = 0; t < T; ++t) {
    const int cur = t & 1;
    if (t + 1 < T) {  // overlap next tile copy with this tile's WMMAs
      const int nxt = (t + 1) & 1;
      const int kn = (t + 1) * 32;
      async_copy_b128(&lA[nxt][lofs],     gA + kn);
      async_copy_b128(&lA[nxt][lofs + 8], gA + kn + 8);
      async_copy_b128(&lB[nxt][lofs],     gB + kn);
      async_copy_b128(&lB[nxt][lofs + 8], gB + kn + 8);
    }

    v16bf bfr[4];
#pragma unroll
    for (int j = 0; j < 4; ++j)
      bfr[j] = load_frag_b(&lB[cur][(wn + j * 16 + l16) * GP + 8 * half]);
#pragma unroll
    for (int i = 0; i < 2; ++i) {
      v16bf af = load_frag_a(&lA[cur][(wm + i * 16 + l16) * GP + 8 * half]);
#pragma unroll
      for (int j = 0; j < 4; ++j) acc[i][j] = wmma_bf16(af, bfr[j], acc[i][j]);
    }

    async_wait0();    // own copies for tile t+1 landed
    __syncthreads();  // all waves' copies visible; all done reading buf[cur]
  }

  // epilogue: route into q / k / v^T
#pragma unroll
  for (int i = 0; i < 2; ++i) {
#pragma unroll
    for (int j = 0; j < 4; ++j) {
#pragma unroll
      for (int v = 0; v < 8; ++v) {
        float val = acc[i][j][v];
        int mg = m0 + wm + i * 16 + v + 8 * half;   // 0..8191
        int d  = n0 + wn + j * 16 + l16;            // 0..2303
        int bi = mg >> 11, wq = mg & 2047;
        int which = d / C_DIM, c = d % C_DIM;
        int h = c >> 8, hd = c & 255;
        long bhw = ((long)(bi * H_DIM + h) * W_LEN + wq) * K_DIM + hd;
        if (which == 0)      qb[bhw] = (bf16)(val * 0.0625f);  // 1/sqrt(256)
        else if (which == 1) kb[bhw] = (bf16)val;
        else vt[((long)(bi * H_DIM + h) * K_DIM + hd) * W_LEN + wq] = (bf16)val;
      }
    }
  }
}

// ---------------- Flash attention (causal, online softmax) -----------------
// One wave owns 16 query rows. KV processed in blocks of 64.
// K/V B-fragments come straight from global (L2-resident on MI455X: 192 MB).
// LDS used only to relayout P from C-layout to A-fragment layout.

#define PPITCH 72  // 64 + 8 pad, keeps 16B alignment (72*2=144 bytes)

__global__ __launch_bounds__(256) void attn_kernel(
    const bf16* __restrict__ qbuf, const bf16* __restrict__ kbuf,
    const bf16* __restrict__ vtbuf, bf16* __restrict__ ctx) {
  __shared__ bf16 pscr[8 * 16 * PPITCH];

  const int tid  = threadIdx.x;
  const int lane = tid & 31, wv = tid >> 5;
  const int half = lane >> 4, l16 = lane & 15;
  const int bh = blockIdx.y;                       // 0..11
  const int q0 = blockIdx.x * 128 + wv * 16;       // wave's query base row

  const bf16* Q  = qbuf  + (long)bh * W_LEN * K_DIM;
  const bf16* Kp = kbuf  + (long)bh * W_LEN * K_DIM;
  const bf16* Vt = vtbuf + (long)bh * K_DIM * W_LEN;
  bf16* pl = pscr + wv * 16 * PPITCH;

  v8f o[16];
#pragma unroll
  for (int t = 0; t < 16; ++t) o[t] = (v8f)(0.0f);
  float mrun[8], lrun[8];
#pragma unroll
  for (int v = 0; v < 8; ++v) { mrun[v] = -__builtin_inff(); lrun[v] = 0.0f; }

  const int jmax = (q0 + 15) >> 6;
  for (int j = 0; j <= jmax; ++j) {
    const int n0 = j * 64;

    // ---- S = Q(16x256) @ K^T(256x64), 4 accum tiles, K-loop of 8 ----
    v8f s[4];
#pragma unroll
    for (int t = 0; t < 4; ++t) s[t] = (v8f)(0.0f);
#pragma unroll
    for (int kc = 0; kc < 8; ++kc) {
      v16bf a = load_frag_a(Q + (long)(q0 + l16) * K_DIM + kc * 32 + 8 * half);
#pragma unroll
      for (int t = 0; t < 4; ++t) {
        v16bf b = load_frag_b(Kp + (long)(n0 + t * 16 + l16) * K_DIM +
                              kc * 32 + 8 * half);
        s[t] = wmma_bf16(a, b, s[t]);
      }
    }

    // ---- causal mask + online softmax (per-row state, rows = v+8*half) ----
#pragma unroll
    for (int v = 0; v < 8; ++v) {
      const int qg = q0 + v + 8 * half;
      float rm = -__builtin_inff();
#pragma unroll
      for (int t = 0; t < 4; ++t) {
        int ng = n0 + t * 16 + l16;
        if (ng > qg) s[t][v] = -__builtin_inff();
        rm = fmaxf(rm, s[t][v]);
      }
#pragma unroll
      for (int xm = 1; xm < 16; xm <<= 1)
        rm = fmaxf(rm, __shfl_xor(rm, xm, 32));
      float mnew = fmaxf(mrun[v], rm);
      float alpha = __expf(mrun[v] - mnew);
      lrun[v] *= alpha;
#pragma unroll
      for (int t = 0; t < 16; ++t) o[t][v] *= alpha;
      float rs = 0.0f;
#pragma unroll
      for (int t = 0; t < 4; ++t) {
        float p = __expf(s[t][v] - mnew);
        s[t][v] = p;
        rs += p;
      }
#pragma unroll
      for (int xm = 1; xm < 16; xm <<= 1) rs += __shfl_xor(rs, xm, 32);
      lrun[v] += rs;
      mrun[v] = mnew;
    }

    // ---- relayout P (C-layout) -> A-fragment layout via per-wave LDS ----
#pragma unroll
    for (int t = 0; t < 4; ++t)
#pragma unroll
      for (int v = 0; v < 8; ++v)
        pl[(v + 8 * half) * PPITCH + t * 16 + l16] = (bf16)s[t][v];
    asm volatile("s_wait_dscnt 0" ::: "memory");  // same-wave DS RAW

    // ---- O += P(16x64) @ V(64x256); V^T rows give contiguous B-frags ----
#pragma unroll
    for (int kc = 0; kc < 2; ++kc) {
      v16bf pa = load_frag_a(pl + l16 * PPITCH + kc * 32 + 8 * half);
#pragma unroll
      for (int t = 0; t < 16; ++t) {
        v16bf b = load_frag_b(Vt + (long)(t * 16 + l16) * W_LEN + n0 +
                              kc * 32 + 8 * half);
        o[t] = wmma_bf16(pa, b, o[t]);
      }
    }
  }

  // ---- normalize and write context [b][w][c] as bf16 ----
  const int bi = bh / H_DIM, h = bh % H_DIM;
#pragma unroll
  for (int v = 0; v < 8; ++v) {
    const int qg = q0 + v + 8 * half;
    float inv = 1.0f / lrun[v];
#pragma unroll
    for (int t = 0; t < 16; ++t)
      ctx[((long)(bi * W_LEN + qg)) * C_DIM + h * K_DIM + t * 16 + l16] =
          (bf16)(o[t][v] * inv);
  }
}

// ---------------- Output projection GEMM ------------------------------------
// out[m][n] = sum_k ctx[m][k] * Wproj[n][k]   (M=8192, N=768, K=768) -> f32

__global__ __launch_bounds__(256) void proj_gemm_kernel(
    const bf16* __restrict__ Xc,   // [8192][768] bf16 context
    const bf16* __restrict__ Wp,   // [768][768] bf16
    float* __restrict__ out) {     // [8192][768] f32
  __shared__ bf16 lA[2][128 * GP];
  __shared__ bf16 lB[2][128 * GP];

  const int tid  = threadIdx.x;
  const int lane = tid & 31, wv = tid >> 5;
  const int half = lane >> 4, l16 = lane & 15;
  const int m0 = blockIdx.x * 128;
  const int n0 = blockIdx.y * 128;
  const int wm = (wv & 3) * 32;
  const int wn = (wv >> 2) * 64;

  v8f acc[2][4];
#pragma unroll
  for (int i = 0; i < 2; ++i)
#pragma unroll
    for (int j = 0; j < 4; ++j) acc[i][j] = (v8f)(0.0f);

  const int lrow = tid >> 1;
  const int lcol = (tid & 1) * 16;
  const bf16* gA = Xc + (long)(m0 + lrow) * C_DIM + lcol;
  const bf16* gB = Wp + (long)(n0 + lrow) * C_DIM + lcol;
  const int lofs = lrow * GP + lcol;

  const int T = C_DIM / 32;

  async_copy_b128(&lA[0][lofs],     gA);
  async_copy_b128(&lA[0][lofs + 8], gA + 8);
  async_copy_b128(&lB[0][lofs],     gB);
  async_copy_b128(&lB[0][lofs + 8], gB + 8);
  async_wait0();
  __syncthreads();

  for (int t = 0; t < T; ++t) {
    const int cur = t & 1;
    if (t + 1 < T) {
      const int nxt = (t + 1) & 1;
      const int kn = (t + 1) * 32;
      async_copy_b128(&lA[nxt][lofs],     gA + kn);
      async_copy_b128(&lA[nxt][lofs + 8], gA + kn + 8);
      async_copy_b128(&lB[nxt][lofs],     gB + kn);
      async_copy_b128(&lB[nxt][lofs + 8], gB + kn + 8);
    }

    v16bf bfr[4];
#pragma unroll
    for (int j = 0; j < 4; ++j)
      bfr[j] = load_frag_b(&lB[cur][(wn + j * 16 + l16) * GP + 8 * half]);
#pragma unroll
    for (int i = 0; i < 2; ++i) {
      v16bf af = load_frag_a(&lA[cur][(wm + i * 16 + l16) * GP + 8 * half]);
#pragma unroll
      for (int j = 0; j < 4; ++j) acc[i][j] = wmma_bf16(af, bfr[j], acc[i][j]);
    }

    async_wait0();
    __syncthreads();
  }

#pragma unroll
  for (int i = 0; i < 2; ++i)
#pragma unroll
    for (int j = 0; j < 4; ++j)
#pragma unroll
      for (int v = 0; v < 8; ++v) {
        int mg = m0 + wm + i * 16 + v + 8 * half;
        int ng = n0 + wn + j * 16 + l16;
        out[(long)mg * C_DIM + ng] = acc[i][j][v];
      }
}

// ---------------- host launch ------------------------------------------------

extern "C" void kernel_launch(void* const* d_in, const int* in_sizes, int n_in,
                              void* d_out, int out_size, void* d_ws,
                              size_t ws_size, hipStream_t stream) {
  (void)in_sizes; (void)n_in; (void)out_size; (void)ws_size;
  const float* X     = (const float*)d_in[0];  // [4,2048,768]
  const float* Wqkv  = (const float*)d_in[1];  // [2304,768]
  const float* Wproj = (const float*)d_in[2];  // [768,768]
  float* out = (float*)d_out;

  char* ws = (char*)d_ws;
  size_t off = 0;
  auto alloc = [&](size_t bytes) -> void* {
    void* p = ws + off;
    off += (bytes + 255) & ~(size_t)255;
    return p;
  };
  const size_t NX  = (size_t)M_ROWS * C_DIM;                 // 6291456
  const size_t NWQ = (size_t)3 * C_DIM * C_DIM;              // 1769472
  const size_t NWP = (size_t)C_DIM * C_DIM;                  // 589824
  const size_t NQK = (size_t)B_DIM * H_DIM * W_LEN * K_DIM;  // 6291456

  bf16* Xbf  = (bf16*)alloc(NX * 2);
  bf16* Wqb  = (bf16*)alloc(NWQ * 2);
  bf16* Wpb  = (bf16*)alloc(NWP * 2);
  bf16* qb   = (bf16*)alloc(NQK * 2);
  bf16* kb   = (bf16*)alloc(NQK * 2);
  bf16* vt   = (bf16*)alloc(NQK * 2);
  bf16* ctx  = (bf16*)alloc(NX * 2);

  cvt_f32_bf16_kernel<<<2048, 256, 0, stream>>>((const float4*)X, Xbf,
                                                (int)(NX / 4));
  cvt_f32_bf16_kernel<<<1024, 256, 0, stream>>>((const float4*)Wqkv, Wqb,
                                                (int)(NWQ / 4));
  cvt_f32_bf16_kernel<<<512, 256, 0, stream>>>((const float4*)Wproj, Wpb,
                                               (int)(NWP / 4));

  qkv_gemm_kernel<<<dim3(M_ROWS / 128, (3 * C_DIM) / 128), 256, 0, stream>>>(
      Xbf, Wqb, qb, kb, vt);

  attn_kernel<<<dim3(W_LEN / 128, B_DIM * H_DIM), 256, 0, stream>>>(qb, kb, vt,
                                                                    ctx);

  proj_gemm_kernel<<<dim3(M_ROWS / 128, C_DIM / 128), 256, 0, stream>>>(
      ctx, Wpb, out);
}
